// EMLAttractorMemory_79663053406678
// MI455X (gfx1250) — matrix operand order repaired and checked by reading, not compiled, not run on hardware
//
#include <hip/hip_runtime.h>

// EMLAttractorMemory forward for MI455X (gfx1250, wave32, WMMA + async-LDS DMA).
//
// Input order assumes jax.tree_util leaf order (dict keys sorted):
//  0 states (B,N,D) f32        1 padding_mask (B,N) bool(u8)
//  2 an_b  3 an_g  4 attractors(A,D)
//  5 drive.b1(512) 6 drive.b2(1) 7 drive.w1(1024,512) 8 drive.w2(512)
//  9 e_bias(1)
// 10 oldconf.b1 11 oldconf.b2 12 oldconf.w1(256,512) 13 oldconf.w2(512)
// 14 on_b 15 on_g
// 16 resist.b1 17 resist.b2 18 resist.w1(1024,512) 19 resist.w2(512)
// 20 sn_b 21 sn_g 22 vp_b(256) 23 vp_w(256,256)
//
// Output (flat f32): attractor_states[16384] | weights[131072] |
//                    null_weight[64] | message[16384] | gate[64]

#define Bv 16
#define Nv 2048
#define Dv 256
#define Hv 512
#define Av 4

#define OFF_ATT  0
#define OFF_W    16384
#define OFF_NULL 147456
#define OFF_MSG  147520
#define OFF_GATE 163904

typedef __attribute__((ext_vector_type(16))) __bf16 v16bf;
typedef __attribute__((ext_vector_type(8)))  float  v8f;

union ABFrag { v16bf v; unsigned int u[8]; };

// Async DMA: global -> LDS, 16B per lane, tracked by ASYNCcnt.
// INST_OFFSET is added to BOTH the LDS and global addresses (ISA 15.18.3),
// so one LDS base + one 32-bit global voffset covers a 64B contiguous run.
#define ASYNC_B128(ldsoff, voff, sbase, imm)                                    \
    asm volatile("global_load_async_to_lds_b128 %0, %1, %2 offset:" #imm       \
                 :: "v"(ldsoff), "v"(voff), "s"(sbase) : "memory")

__device__ __forceinline__ unsigned short f2bf(float f) {
    union { float f; unsigned int u; } c; c.f = f;
    unsigned int u = c.u;
    u += 0x7fffu + ((u >> 16) & 1u);
    return (unsigned short)(u >> 16);
}

// CDNA5 native transcendental tanh (TRANS32 class) — avoids libm's branchy expf.
__device__ __forceinline__ float tanh_hw(float x) {
    float r;
    asm volatile("v_tanh_f32 %0, %1" : "=v"(r) : "v"(x));
    return r;
}

__device__ __forceinline__ float gelu_tanh(float x) {
    float x3 = x * x * x;
    float t  = 0.7978845608028654f * (x + 0.044715f * x3);
    return 0.5f * x * (1.0f + tanh_hw(t));
}

__device__ __forceinline__ float wave_sum(float v) {
#pragma unroll
    for (int o = 16; o > 0; o >>= 1) v += __shfl_xor(v, o);
    return v;
}

__device__ __forceinline__ float blk_sum(float v, float* red) {
    int t = threadIdx.x;
    red[t] = v; __syncthreads();
#pragma unroll
    for (int s = 128; s > 0; s >>= 1) { if (t < s) red[t] += red[t + s]; __syncthreads(); }
    float r = red[0]; __syncthreads();
    return r;
}

__device__ __forceinline__ float blk_max(float v, float* red) {
    int t = threadIdx.x;
    red[t] = v; __syncthreads();
#pragma unroll
    for (int s = 128; s > 0; s >>= 1) { if (t < s) red[t] = fmaxf(red[t], red[t + s]); __syncthreads(); }
    float r = red[0]; __syncthreads();
    return r;
}

// ---------------- K1: layernorm(states) -> bf16 ns (row-major) ----------------
__global__ __launch_bounds__(256) void k_ln_pack(
    const float* __restrict__ states, const float* __restrict__ g,
    const float* __restrict__ b, unsigned short* __restrict__ ns) {
    int wave = threadIdx.x >> 5, lane = threadIdx.x & 31;
    long row = (long)blockIdx.x * 8 + wave;
    int c0 = lane * 8;
    const float* x = states + row * Dv + c0;
    float xv[8];
    float4 p0 = *(const float4*)(x);
    float4 p1 = *(const float4*)(x + 4);
    xv[0]=p0.x; xv[1]=p0.y; xv[2]=p0.z; xv[3]=p0.w;
    xv[4]=p1.x; xv[5]=p1.y; xv[6]=p1.z; xv[7]=p1.w;
    float s = 0.f;
#pragma unroll
    for (int i = 0; i < 8; ++i) s += xv[i];
    float m = wave_sum(s) * (1.0f / Dv);
    float d2 = 0.f;
#pragma unroll
    for (int i = 0; i < 8; ++i) { float d = xv[i] - m; d2 += d * d; }
    float var = wave_sum(d2) * (1.0f / Dv);
    float sc = rsqrtf(var + 1e-5f);
    unsigned int pk[4];
#pragma unroll
    for (int i = 0; i < 4; ++i) {
        float y0 = (xv[2*i]   - m) * sc * g[c0 + 2*i]   + b[c0 + 2*i];
        float y1 = (xv[2*i+1] - m) * sc * g[c0 + 2*i+1] + b[c0 + 2*i+1];
        pk[i] = (unsigned int)f2bf(y0) | ((unsigned int)f2bf(y1) << 16);
    }
    *(uint4*)(ns + row * Dv + c0) = make_uint4(pk[0], pk[1], pk[2], pk[3]);
}

// -------- K2: LN(attractors), old_conf, per-attractor bias rows rd/rr --------
__global__ __launch_bounds__(256) void k_prep(
    const float* __restrict__ attractors, const float* __restrict__ an_g, const float* __restrict__ an_b,
    const float* __restrict__ ocw1, const float* __restrict__ ocb1,
    const float* __restrict__ ocw2, const float* __restrict__ ocb2,
    const float* __restrict__ w1d, const float* __restrict__ b1d,
    const float* __restrict__ w1r, const float* __restrict__ b1r,
    float* __restrict__ na, float* __restrict__ oldconf,
    float* __restrict__ rdo, float* __restrict__ rro) {
    __shared__ float naS[Av * Dv];
    __shared__ float red[256];
    int t = threadIdx.x, wave = t >> 5, lane = t & 31;
    if (wave < Av) {
        int a = wave, c0 = lane * 8;
        const float* x = attractors + a * Dv + c0;
        float xv[8];
        float4 p0 = *(const float4*)(x);
        float4 p1 = *(const float4*)(x + 4);
        xv[0]=p0.x; xv[1]=p0.y; xv[2]=p0.z; xv[3]=p0.w;
        xv[4]=p1.x; xv[5]=p1.y; xv[6]=p1.z; xv[7]=p1.w;
        float s = 0.f;
#pragma unroll
        for (int i = 0; i < 8; ++i) s += xv[i];
        float m = wave_sum(s) * (1.0f / Dv);
        float d2 = 0.f;
#pragma unroll
        for (int i = 0; i < 8; ++i) { float d = xv[i] - m; d2 += d * d; }
        float var = wave_sum(d2) * (1.0f / Dv);
        float sc = rsqrtf(var + 1e-5f);
#pragma unroll
        for (int i = 0; i < 8; ++i) {
            int c = c0 + i;
            float y = (xv[i] - m) * sc * an_g[c] + an_b[c];
            naS[a * Dv + c] = y;
            na[a * Dv + c] = y;
        }
    }
    __syncthreads();
    for (int a = 0; a < Av; ++a) {
        float partial = 0.f;
        for (int hh = t; hh < Hv; hh += 256) {
            float dot = 0.f;
            for (int d = 0; d < Dv; ++d) dot = fmaf(naS[a * Dv + d], ocw1[(long)d * Hv + hh], dot);
            partial = fmaf(gelu_tanh(dot + ocb1[hh]), ocw2[hh], partial);
        }
        float tot = blk_sum(partial, red);
        if (t == 0) oldconf[a] = tot + ocb2[0];
    }
    for (int a = 0; a < Av; ++a) {
        for (int hh = t; hh < Hv; hh += 256) {
            float sd = 0.f, sr = 0.f;
            for (int d = 0; d < Dv; ++d) {
                float nv = naS[a * Dv + d];
                sd = fmaf(nv, w1d[(long)d * Hv + hh] + w1d[(long)(768 + d) * Hv + hh], sd);
                sr = fmaf(nv, w1r[(long)d * Hv + hh] + w1r[(long)(768 + d) * Hv + hh], sr);
            }
            rdo[a * Hv + hh] = sd + b1d[hh];
            rro[a * Hv + hh] = sr + b1r[hh];
        }
    }
}

// ---- K3: W_a^T = (diag(na_a)w1_c + (w1_e - w1_s))^T as bf16 [a][h][k]; vpW^T ----
__global__ __launch_bounds__(256) void k_build_w(
    const float* __restrict__ na, const float* __restrict__ w1d, const float* __restrict__ w1r,
    const float* __restrict__ vp_w,
    unsigned short* __restrict__ wdt, unsigned short* __restrict__ wrt,
    unsigned short* __restrict__ vpwt) {
    long idx = (long)blockIdx.x * 256 + threadIdx.x;
    if (idx < 524288) {
        int k = (int)(idx & 255), h = (int)((idx >> 8) & 511), a = (int)(idx >> 17);
        float v = na[a * Dv + k] * w1d[(long)(512 + k) * Hv + h]
                + (w1d[(long)(256 + k) * Hv + h] - w1d[(long)(768 + k) * Hv + h]);
        wdt[idx] = f2bf(v);
    } else if (idx < 1048576) {
        long j = idx - 524288;
        int k = (int)(j & 255), h = (int)((j >> 8) & 511), a = (int)(j >> 17);
        float v = na[a * Dv + k] * w1r[(long)(512 + k) * Hv + h]
                + (w1r[(long)(256 + k) * Hv + h] - w1r[(long)(768 + k) * Hv + h]);
        wrt[j] = f2bf(v);
    } else if (idx < 1114112) {
        long j = idx - 1048576;
        int k = (int)(j & 255), n = (int)(j >> 8);
        vpwt[j] = f2bf(vp_w[(long)k * Dv + n]);
    }
}

// ------- K4: fused drive/resist GEMM + gelu + dot(w2) -> energy -------
// 16-chunk stream (8 drive + 8 resist), LDS double-buffered, async DMA fills
// chunk c+1 while WMMAs consume chunk c.
__global__ __launch_bounds__(256) void k_energy(
    const unsigned short* __restrict__ ns,
    const unsigned short* __restrict__ wdt, const unsigned short* __restrict__ wrt,
    const float* __restrict__ rd, const float* __restrict__ rr,
    const float* __restrict__ w2d, const float* __restrict__ w2r,
    const float* __restrict__ b2d, const float* __restrict__ b2r,
    const float* __restrict__ ebias, float* __restrict__ energy) {
    __shared__ __align__(16) unsigned short At[2][64 * 32];   // [buf][row][k]
    __shared__ __align__(16) unsigned short Wt[2][Hv * 32];   // [buf][h][k]
    __shared__ float red[2][64];

    const int blk = blockIdx.x;
    const int ntile = blk & 31;
    const int a = (blk >> 5) & 3;
    const int b = blk >> 7;
    const int n0 = ntile * 64;

    const int tid = threadIdx.x;
    const int wave = tid >> 5;
    const int lane = tid & 31;
    const int l16 = lane & 15;
    const int lhi = lane >> 4;

    if (tid < 64) { red[0][tid] = 0.f; red[1][tid] = 0.f; }

    const unsigned short* wdtA = wdt + (long)a * Hv * Dv;
    const unsigned short* wrtA = wrt + (long)a * Hv * Dv;
    const float* rsel[2]  = { rd + a * Hv, rr + a * Hv };
    const float* w2sel[2] = { w2d, w2r };
    const long nsbase = ((long)b * Nv + n0) * Dv;

    const int ar = tid >> 2, ak = (tid & 3) * 8;     // A-tile slice owned by this thread

    auto issue = [&](int c) {
        const int buf = c & 1;
        const int kc  = (c & 7) * 32;
        const unsigned short* wp = (c >> 3) ? wrtA : wdtA;
        // A tile: 64x32 bf16, one b128 per thread
        unsigned aLds = (unsigned)(size_t)&At[buf][ar * 32 + ak];
        unsigned aOff = (unsigned)((nsbase + (long)ar * Dv + kc + ak) * 2);
        ASYNC_B128(aLds, aOff, ns, 0);
        // W tile: [h][k] 512x32 bf16, rows tid and tid+256 (4 b128 each)
        unsigned wLds0 = (unsigned)(size_t)&Wt[buf][tid * 32];
        unsigned wOff0 = (unsigned)(((long)tid * Dv + kc) * 2);
        ASYNC_B128(wLds0, wOff0, wp, 0);
        ASYNC_B128(wLds0, wOff0, wp, 16);
        ASYNC_B128(wLds0, wOff0, wp, 32);
        ASYNC_B128(wLds0, wOff0, wp, 48);
        unsigned wLds1 = (unsigned)(size_t)&Wt[buf][(tid + 256) * 32];
        unsigned wOff1 = (unsigned)(((long)(tid + 256) * Dv + kc) * 2);
        ASYNC_B128(wLds1, wOff1, wp, 0);
        ASYNC_B128(wLds1, wOff1, wp, 16);
        ASYNC_B128(wLds1, wOff1, wp, 32);
        ASYNC_B128(wLds1, wOff1, wp, 48);
    };

    v8f acc[4][4];
#pragma unroll
    for (int mt = 0; mt < 4; ++mt)
#pragma unroll
        for (int nt = 0; nt < 4; ++nt)
#pragma unroll
            for (int j = 0; j < 8; ++j) acc[mt][nt][j] = 0.f;

    issue(0);
    for (int c = 0; c < 16; ++c) {
        const int buf = c & 1;
        if (c < 15) {
            issue(c + 1);
            asm volatile("s_wait_asynccnt 9" ::: "memory");   // chunk c retired, c+1 in flight
        } else {
            asm volatile("s_wait_asynccnt 0" ::: "memory");
        }
        __syncthreads();

        ABFrag bfr[4];
#pragma unroll
        for (int nt = 0; nt < 4; ++nt) {
            int h = wave * 64 + nt * 16 + l16;          // B: n=l16, k=(t&7)+8*lhi+16*(t>>3)
#pragma unroll
            for (int j = 0; j < 8; ++j)
                bfr[nt].u[j] = *(const unsigned int*)&Wt[buf][h * 32 + 2 * (j & 3) + 8 * lhi + 16 * (j >> 2)];
        }
#pragma unroll
        for (int mt = 0; mt < 4; ++mt) {
            ABFrag af;
            int m = mt * 16 + l16;                       // A: m=l16, k=(t&7)+8*lhi+16*(t>>3)
#pragma unroll
            for (int j = 0; j < 8; ++j) {
                int k0 = 2 * (j & 3) + 8 * lhi + 16 * (j >> 2);
                af.u[j] = *(const unsigned int*)&At[buf][m * 32 + k0];
            }
#pragma unroll
            for (int nt = 0; nt < 4; ++nt)
                acc[mt][nt] = __builtin_amdgcn_wmma_f32_16x16x32_bf16(
                    false, af.v, false, bfr[nt].v, (short)0, acc[mt][nt], false, false);
        }

        if ((c & 7) == 7) {
            // epilogue: gelu(h + r_a) . w2, reduce this wave's 64 hidden cols
            const int pass = c >> 3;
            float part[4][8];
#pragma unroll
            for (int mt = 0; mt < 4; ++mt)
#pragma unroll
                for (int j = 0; j < 8; ++j) part[mt][j] = 0.f;
#pragma unroll
            for (int nt = 0; nt < 4; ++nt) {
                int h = wave * 64 + nt * 16 + l16;
                float rv = rsel[pass][h];
                float wv = w2sel[pass][h];
#pragma unroll
                for (int mt = 0; mt < 4; ++mt)
#pragma unroll
                    for (int j = 0; j < 8; ++j)
                        part[mt][j] += gelu_tanh(acc[mt][nt][j] + rv) * wv;
            }
#pragma unroll
            for (int mt = 0; mt < 4; ++mt)
#pragma unroll
                for (int j = 0; j < 8; ++j) {
                    float p = part[mt][j];
#pragma unroll
                    for (int o = 1; o < 16; o <<= 1) p += __shfl_xor(p, o);
                    part[mt][j] = p;
                }
            if (l16 == 0) {
#pragma unroll
                for (int mt = 0; mt < 4; ++mt)
#pragma unroll
                    for (int j = 0; j < 8; ++j)
                        atomicAdd(&red[pass][mt * 16 + j + 8 * lhi], part[mt][j]);
            }
#pragma unroll
            for (int mt = 0; mt < 4; ++mt)
#pragma unroll
                for (int nt = 0; nt < 4; ++nt)
#pragma unroll
                    for (int j = 0; j < 8; ++j) acc[mt][nt][j] = 0.f;
        }
        __syncthreads();   // protect buffer reuse by next issue
    }

    if (tid < 64) {
        float drive  = red[0][tid] + b2d[0];
        float rin    = red[1][tid] + b2r[0];
        float resist = fmaxf(rin, 0.f) + log1pf(__expf(-fabsf(rin)));   // softplus
        float raw    = drive - resist + ebias[0];
        float e      = fminf(fmaxf(raw, -3.0f), 3.0f);
        energy[((long)(b * Av + a)) * Nv + n0 + tid] = e;
    }
}

// -------------------- K5: values = ns @ vp_w + vp_b --------------------
__global__ __launch_bounds__(256) void k_values(
    const unsigned short* __restrict__ ns, const unsigned short* __restrict__ vpwt,
    const float* __restrict__ vp_b, float* __restrict__ values) {
    __shared__ __align__(16) unsigned short At[2][64 * 32];
    __shared__ __align__(16) unsigned short Wt[2][Dv * 32];
    const int tid = threadIdx.x;
    const int wave = tid >> 5, lane = tid & 31, l16 = lane & 15, lhi = lane >> 4;
    const long rowbase = (long)blockIdx.x * 64;
    const int ar = tid >> 2, ak = (tid & 3) * 8;

    auto issue = [&](int c) {
        const int buf = c & 1;
        const int kc  = c * 32;
        unsigned aLds = (unsigned)(size_t)&At[buf][ar * 32 + ak];
        unsigned aOff = (unsigned)(((rowbase + ar) * Dv + kc + ak) * 2);
        ASYNC_B128(aLds, aOff, ns, 0);
        unsigned wLds = (unsigned)(size_t)&Wt[buf][tid * 32];
        unsigned wOff = (unsigned)(((long)tid * Dv + kc) * 2);
        ASYNC_B128(wLds, wOff, vpwt, 0);
        ASYNC_B128(wLds, wOff, vpwt, 16);
        ASYNC_B128(wLds, wOff, vpwt, 32);
        ASYNC_B128(wLds, wOff, vpwt, 48);
    };

    v8f acc[4][2];
#pragma unroll
    for (int mt = 0; mt < 4; ++mt)
#pragma unroll
        for (int nt = 0; nt < 2; ++nt)
#pragma unroll
            for (int j = 0; j < 8; ++j) acc[mt][nt][j] = 0.f;

    issue(0);
    for (int c = 0; c < 8; ++c) {
        const int buf = c & 1;
        if (c < 7) {
            issue(c + 1);
            asm volatile("s_wait_asynccnt 5" ::: "memory");
        } else {
            asm volatile("s_wait_asynccnt 0" ::: "memory");
        }
        __syncthreads();
        ABFrag bfr[2];
#pragma unroll
        for (int nt = 0; nt < 2; ++nt) {
            int h = wave * 32 + nt * 16 + l16;
#pragma unroll
            for (int j = 0; j < 8; ++j)
                bfr[nt].u[j] = *(const unsigned int*)&Wt[buf][h * 32 + 2 * (j & 3) + 8 * lhi + 16 * (j >> 2)];
        }
#pragma unroll
        for (int mt = 0; mt < 4; ++mt) {
            ABFrag af;
            int m = mt * 16 + l16;
#pragma unroll
            for (int j = 0; j < 8; ++j) {
                int k0 = 2 * (j & 3) + 8 * lhi + 16 * (j >> 2);
                af.u[j] = *(const unsigned int*)&At[buf][m * 32 + k0];
            }
#pragma unroll
            for (int nt = 0; nt < 2; ++nt)
                acc[mt][nt] = __builtin_amdgcn_wmma_f32_16x16x32_bf16(
                    false, af.v, false, bfr[nt].v, (short)0, acc[mt][nt], false, false);
        }
        __syncthreads();
    }
#pragma unroll
    for (int mt = 0; mt < 4; ++mt)
#pragma unroll
        for (int nt = 0; nt < 2; ++nt) {
            int colg = wave * 32 + nt * 16 + l16;
            float bias = vp_b[colg];
#pragma unroll
            for (int j = 0; j < 8; ++j) {
                long rowg = rowbase + mt * 16 + j + 8 * lhi;
                values[rowg * Dv + colg] = acc[mt][nt][j] + bias;
            }
        }
}

// -------- K6: softmax over N (+null) -> weights, null_weight, gate --------
__global__ __launch_bounds__(256) void k_softmax(
    const float* __restrict__ energy, const unsigned char* __restrict__ mask,
    const float* __restrict__ oldconf, float* __restrict__ out) {
    __shared__ float red[256];
    int ba = blockIdx.x, b = ba >> 2, a = ba & 3, t = threadIdx.x;
    float ev[8], me[8];
    float mx = -1e30f;
#pragma unroll
    for (int i = 0; i < 8; ++i) {
        int n = t + 256 * i;
        float e = energy[(long)ba * Nv + n];
        ev[i] = e;
        me[i] = (mask[(long)b * Nv + n] != 0) ? e : -1e9f;
        mx = fmaxf(mx, me[i]);
    }
    mx = fmaxf(blk_max(mx, red), 0.0f);       // null logit = 0
    float w[8], s = 0.f;
#pragma unroll
    for (int i = 0; i < 8; ++i) { w[i] = __expf(me[i] - mx); s += w[i]; }
    s = blk_sum(s, red);
    float total = s + __expf(-mx);
    float inv = 1.0f / total;
    float ne = 0.f;
#pragma unroll
    for (int i = 0; i < 8; ++i) {
        float wi = w[i] * inv;
        out[OFF_W + (long)ba * Nv + t + 256 * i] = wi;
        ne += wi * ev[i];
    }
    ne = blk_sum(ne, red);
    if (t == 0) {
        float nullw = __expf(-mx) * inv;
        out[OFF_NULL + ba] = nullw;
        float g = (1.0f / (1.0f + __expf(-(ne - oldconf[a])))) * (1.0f - nullw);
        out[OFF_GATE + ba] = g;
    }
}

// ---- K7: message = weights@values; gated update; output LN -> attractor_states ----
__global__ __launch_bounds__(256) void k_message(
    const float* __restrict__ values, const float* __restrict__ attractors,
    const float* __restrict__ on_g, const float* __restrict__ on_b,
    float* __restrict__ out) {
    __shared__ float wch[256];
    __shared__ float red[256];
    int ba = blockIdx.x, b = ba >> 2, a = ba & 3, t = threadIdx.x;
    float msg = 0.f;
    for (int n0 = 0; n0 < Nv; n0 += 256) {
        __syncthreads();
        wch[t] = out[OFF_W + (long)ba * Nv + n0 + t];
        __syncthreads();
        const float* vb = values + ((long)b * Nv + n0) * Dv + t;
#pragma unroll 8
        for (int i = 0; i < 256; ++i) msg = fmaf(wch[i], vb[(long)i * Dv], msg);
    }
    out[OFF_MSG + ba * Dv + t] = msg;
    float g = out[OFF_GATE + ba];
    float attr = attractors[a * Dv + t];
    float nsv = attr + g * (msg - attr);
    float m = blk_sum(nsv, red) * (1.0f / Dv);
    float dv = nsv - m;
    float var = blk_sum(dv * dv, red) * (1.0f / Dv);
    out[OFF_ATT + ba * Dv + t] = dv * rsqrtf(var + 1e-5f) * on_g[t] + on_b[t];
}

extern "C" void kernel_launch(void* const* d_in, const int* in_sizes, int n_in,
                              void* d_out, int out_size, void* d_ws, size_t ws_size,
                              hipStream_t stream) {
    (void)in_sizes; (void)n_in; (void)out_size; (void)ws_size;
    const float* states          = (const float*)d_in[0];
    const unsigned char* mask    = (const unsigned char*)d_in[1];
    const float* an_b            = (const float*)d_in[2];
    const float* an_g            = (const float*)d_in[3];
    const float* attractors      = (const float*)d_in[4];
    const float* d_b1            = (const float*)d_in[5];
    const float* d_b2            = (const float*)d_in[6];
    const float* d_w1            = (const float*)d_in[7];
    const float* d_w2            = (const float*)d_in[8];
    const float* e_bias          = (const float*)d_in[9];
    const float* oc_b1           = (const float*)d_in[10];
    const float* oc_b2           = (const float*)d_in[11];
    const float* oc_w1           = (const float*)d_in[12];
    const float* oc_w2           = (const float*)d_in[13];
    const float* on_b            = (const float*)d_in[14];
    const float* on_g            = (const float*)d_in[15];
    const float* r_b1            = (const float*)d_in[16];
    const float* r_b2            = (const float*)d_in[17];
    const float* r_w1            = (const float*)d_in[18];
    const float* r_w2            = (const float*)d_in[19];
    const float* sn_b            = (const float*)d_in[20];
    const float* sn_g            = (const float*)d_in[21];
    const float* vp_b            = (const float*)d_in[22];
    const float* vp_w            = (const float*)d_in[23];

    char* ws = (char*)d_ws;
    unsigned short* ns   = (unsigned short*)(ws + 0);          // 16,777,216 B
    unsigned short* wdt  = (unsigned short*)(ws + 16777216);   //  1,048,576 B
    unsigned short* wrt  = (unsigned short*)(ws + 17825792);   //  1,048,576 B
    unsigned short* vpwt = (unsigned short*)(ws + 18874368);   //    131,072 B
    float* rd            = (float*)(ws + 19005440);            //      8,192 B
    float* rr            = (float*)(ws + 19013632);            //      8,192 B
    float* na            = (float*)(ws + 19021824);            //      4,096 B
    float* oldconf       = (float*)(ws + 19025920);            //        256 B
    float* energybuf     = (float*)(ws + 19026176);            //    524,288 B
    float* valuesbuf     = (float*)(ws + 19550464);            // 33,554,432 B
    float* out           = (float*)d_out;

    k_ln_pack <<<4096, 256, 0, stream>>>(states, sn_g, sn_b, ns);
    k_prep    <<<1,    256, 0, stream>>>(attractors, an_g, an_b, oc_w1, oc_b1, oc_w2, oc_b2,
                                         d_w1, d_b1, r_w1, r_b1, na, oldconf, rd, rr);
    k_build_w <<<4352, 256, 0, stream>>>(na, d_w1, r_w1, vp_w, wdt, wrt, vpwt);
    k_energy  <<<2048, 256, 0, stream>>>(ns, wdt, wrt, rd, rr, d_w2, r_w2, d_b2, r_b2,
                                         e_bias, energybuf);
    k_values  <<<512,  256, 0, stream>>>(ns, vpwt, vp_b, valuesbuf);
    k_softmax <<<64,   256, 0, stream>>>(energybuf, mask, oldconf, out);
    k_message <<<64,   256, 0, stream>>>(valuesbuf, attractors, on_g, on_b, out);
}
